// EnhancedTransformerTextClassifier_53618371723532
// MI455X (gfx1250) — matrix operand order repaired and checked by reading, not compiled, not run on hardware
//
#include <hip/hip_runtime.h>
#include <hip/hip_bf16.h>

typedef __attribute__((ext_vector_type(16))) _Float16 v16h;
typedef __attribute__((ext_vector_type(8)))  float    v8f;
typedef _Float16 half_t;

static constexpr int Bb = 32, Ss = 512, Dd = 512, Hh = 8, Ll = 6, FFf = 2048, HDd = 64;
static constexpr int NT = Bb * Ss; // 16384 tokens

__device__ inline v8f zero8() {
  v8f z;
  #pragma unroll
  for (int i = 0; i < 8; ++i) z[i] = 0.f;
  return z;
}

__device__ inline float gelu_exact(float x) {
  return 0.5f * x * (1.0f + erff(x * 0.70710678118654752440f));
}

// ---------------------------------------------------------------- conversion
__global__ void f32_to_f16_kernel(const float* __restrict__ in, half_t* __restrict__ out, size_t n) {
  size_t i = (size_t)blockIdx.x * 256 + threadIdx.x;
  if (i < n) out[i] = (half_t)in[i];
}

// ---------------------------------------------------------------- embedding
__global__ __launch_bounds__(256) void embed_kernel(
    const long long* __restrict__ ids, const float* __restrict__ tok,
    const float* __restrict__ pos, const float* __restrict__ typ,
    float* __restrict__ x, half_t* __restrict__ xh) {
  size_t idx = (size_t)blockIdx.x * 256 + threadIdx.x; // over NT*Dd
  int d = (int)(idx & (Dd - 1));
  size_t n = idx >> 9;          // /Dd
  int s = (int)(n & (Ss - 1));  // token position
  float v = tok[(size_t)ids[n] * Dd + d] + pos[(size_t)s * Dd + d] + typ[d];
  x[idx] = v; xh[idx] = (half_t)v;
}

// ---------------------------------------------------------------- WMMA GEMM
// Y[M,Ncols] = A[M,K](f16) @ W[K,Ncols](f16) + bias ; optional exact GELU.
// Block tile 128x128, 8 waves in 4(M) x 2(N) grid, K chunks of 32 via LDS.
__global__ __launch_bounds__(256) void gemm_wmma_kernel(
    const half_t* __restrict__ A, const half_t* __restrict__ W,
    const float* __restrict__ bias, float* __restrict__ outF,
    half_t* __restrict__ outH, int M, int K, int Ncols, int act) {
  __shared__ alignas(32) half_t Ash[128 * 32];   // [row][k]   row-stride 32
  __shared__ alignas(32) half_t Bts[128 * 32];   // [col][k]   col-stride 32 (transposed)
  int t = threadIdx.x, lane = t & 31, w = t >> 5;
  int lo16 = lane & 15, hi = lane >> 4;          // hi in {0,1}
  int wN = w & 1, wM = w >> 1;                   // wave tile: 32 rows x 64 cols
  int nBase = blockIdx.x * 128, mBase = blockIdx.y * 128;

  v8f acc[2][4];
  #pragma unroll
  for (int mt = 0; mt < 2; ++mt)
    #pragma unroll
    for (int nt = 0; nt < 4; ++nt) acc[mt][nt] = zero8();

  for (int kc = 0; kc < K; kc += 32) {
    { // cooperative tile loads
      int r = t >> 1, c = (t & 1) * 16;
      *(v16h*)(&Ash[r * 32 + c]) =
          *(const v16h*)(A + (size_t)(mBase + r) * K + kc + c);
      int kk = t >> 3, c2 = (t & 7) * 16;
      v16h wv = *(const v16h*)(W + (size_t)(kc + kk) * Ncols + nBase + c2);
      #pragma unroll
      for (int i = 0; i < 16; ++i) Bts[(c2 + i) * 32 + kk] = wv[i];
    }
    __syncthreads();
    v16h af[2], bf[4];
    #pragma unroll
    for (int mt = 0; mt < 2; ++mt) {
      const half_t* ap = &Ash[(wM * 32 + mt * 16 + lo16) * 32];
      #pragma unroll
      for (int j = 0; j < 8; ++j) { // A layout: per-lane K-contiguous, +16 for hi half-wave
        int kk2 = (hi << 4) + (j << 1);
        af[mt][2 * j] = ap[kk2]; af[mt][2 * j + 1] = ap[kk2 + 1];
      }
    }
    #pragma unroll
    for (int nt = 0; nt < 4; ++nt) // B layout: per-lane K-contiguous, +16 for hi half-wave
      bf[nt] = *(const v16h*)(&Bts[(wN * 64 + nt * 16 + lo16) * 32 + (hi << 4)]);
    #pragma unroll
    for (int mt = 0; mt < 2; ++mt)
      #pragma unroll
      for (int nt = 0; nt < 4; ++nt)
        acc[mt][nt] = __builtin_amdgcn_wmma_f32_16x16x32_f16(
            false, af[mt], false, bf[nt], (short)0, acc[mt][nt], false, false);
    __syncthreads();
  }
  #pragma unroll
  for (int mt = 0; mt < 2; ++mt)
    #pragma unroll
    for (int nt = 0; nt < 4; ++nt) {
      int col = nBase + wN * 64 + nt * 16 + lo16;
      float bv = bias[col];
      #pragma unroll
      for (int r = 0; r < 8; ++r) {
        int row = mBase + wM * 32 + mt * 16 + r + (hi << 3);
        float val = acc[mt][nt][r] + bv;
        if (act) val = gelu_exact(val);
        size_t oidx = (size_t)row * Ncols + col;
        if (outH) outH[oidx] = (half_t)val; else outF[oidx] = val;
      }
    }
}

// ---------------------------------------------------------------- attention
// grid (4, H, B); 8 waves, each wave owns one 16-row query block; flash-style
// online softmax over key chunks of 128 staged in LDS.
__global__ __launch_bounds__(256) void attn_kernel(
    const half_t* __restrict__ q, const half_t* __restrict__ k,
    const half_t* __restrict__ v, const int* __restrict__ mask,
    const float* __restrict__ attn_bias, half_t* __restrict__ out) {
  int qg = blockIdx.x, h = blockIdx.y, b = blockIdx.z;
  int t = threadIdx.x, lane = t & 31, w = t >> 5;
  int lo16 = lane & 15, hi = lane >> 4;

  __shared__ alignas(32) half_t Ksh[128 * 64];     // [key][hd]
  __shared__ alignas(32) half_t Vts[64 * 128];     // [hd][key] (transposed)
  __shared__ alignas(32) half_t Psh[8][16 * 32];   // per-wave P tile

  int qrow0 = (qg * 8 + w) * 16;
  const half_t* qp = q + ((size_t)(b * Ss) + qrow0 + lo16) * Dd + h * HDd;
  v16h a_lo, a_hi;
  #pragma unroll
  for (int j = 0; j < 8; ++j) {
    int kk = (hi << 4) + (j << 1);
    a_lo[2 * j] = qp[kk];      a_lo[2 * j + 1] = qp[kk + 1];
    a_hi[2 * j] = qp[32 + kk]; a_hi[2 * j + 1] = qp[32 + kk + 1];
  }
  float m_[8], l_[8];
  v8f o_[4];
  #pragma unroll
  for (int r = 0; r < 8; ++r) { m_[r] = -1e30f; l_[r] = 0.f; }
  #pragma unroll
  for (int nt = 0; nt < 4; ++nt) o_[nt] = zero8();

  const float scale = 0.125f; // 1/sqrt(64)
  const float ab = attn_bias[h];

  for (int kc = 0; kc < 4; ++kc) {
    { // stage K chunk and V^T chunk
      int r = t >> 1, c = (t & 1) * 32;
      const half_t* kp = k + ((size_t)(b * Ss) + kc * 128 + r) * Dd + h * HDd + c;
      *(v16h*)(&Ksh[r * 64 + c]) = *(const v16h*)kp;
      *(v16h*)(&Ksh[r * 64 + c + 16]) = *(const v16h*)(kp + 16);
      const half_t* vp = v + ((size_t)(b * Ss) + kc * 128 + r) * Dd + h * HDd + c;
      v16h v0 = *(const v16h*)vp, v1 = *(const v16h*)(vp + 16);
      #pragma unroll
      for (int i = 0; i < 16; ++i) {
        Vts[(c + i) * 128 + r] = v0[i];
        Vts[(c + 16 + i) * 128 + r] = v1[i];
      }
    }
    __syncthreads();
    #pragma unroll 1
    for (int tt = 0; tt < 4; ++tt) {
      int kb = tt * 32;
      v8f sc[2];
      #pragma unroll
      for (int st = 0; st < 2; ++st) { // two 16-key score tiles
        int key = kb + st * 16 + lo16;
        v16h b_lo = *(const v16h*)(&Ksh[key * 64 + (hi << 4)]);
        v16h b_hi = *(const v16h*)(&Ksh[key * 64 + 32 + (hi << 4)]);
        v8f c0 = zero8();
        c0 = __builtin_amdgcn_wmma_f32_16x16x32_f16(false, a_lo, false, b_lo, (short)0, c0, false, false);
        c0 = __builtin_amdgcn_wmma_f32_16x16x32_f16(false, a_hi, false, b_hi, (short)0, c0, false, false);
        int mk = mask[b * Ss + kc * 128 + key];
        #pragma unroll
        for (int r = 0; r < 8; ++r) {
          float vv = c0[r] * scale + ab;
          if (mk == 0) vv = -1e9f;
          c0[r] = vv;
        }
        sc[st] = c0;
      }
      float Mn[8], alpha[8];
      #pragma unroll
      for (int r = 0; r < 8; ++r) { // row max across the 16-lane row group
        float rm = fmaxf(sc[0][r], sc[1][r]);
        #pragma unroll
        for (int mskk = 1; mskk < 16; mskk <<= 1) rm = fmaxf(rm, __shfl_xor(rm, mskk, 32));
        Mn[r] = fmaxf(m_[r], rm);
        alpha[r] = __expf(m_[r] - Mn[r]);
      }
      #pragma unroll
      for (int r = 0; r < 8; ++r) {
        float p0 = __expf(sc[0][r] - Mn[r]);
        float p1 = __expf(sc[1][r] - Mn[r]);
        sc[0][r] = p0; sc[1][r] = p1;
        float rs = p0 + p1;
        #pragma unroll
        for (int mskk = 1; mskk < 16; mskk <<= 1) rs += __shfl_xor(rs, mskk, 32);
        l_[r] = l_[r] * alpha[r] + rs;
        m_[r] = Mn[r];
      }
      #pragma unroll
      for (int nt = 0; nt < 4; ++nt)
        #pragma unroll
        for (int r = 0; r < 8; ++r) o_[nt][r] *= alpha[r];
      // C-layout -> A-layout relayout of P through per-wave LDS
      half_t* pw = &Psh[w][0];
      #pragma unroll
      for (int st = 0; st < 2; ++st)
        #pragma unroll
        for (int r = 0; r < 8; ++r)
          pw[(r + (hi << 3)) * 32 + st * 16 + lo16] = (half_t)sc[st][r];
      asm volatile("s_wait_dscnt 0" ::: "memory"); // wave32 lockstep: wait is enough
      v16h pf;
      #pragma unroll
      for (int j = 0; j < 8; ++j) {
        int kk2 = (hi << 4) + (j << 1);
        pf[2 * j] = pw[lo16 * 32 + kk2]; pf[2 * j + 1] = pw[lo16 * 32 + kk2 + 1];
      }
      #pragma unroll
      for (int nt = 0; nt < 4; ++nt) {
        int hd = nt * 16 + lo16;
        v16h vf = *(const v16h*)(&Vts[hd * 128 + kb + (hi << 4)]);
        o_[nt] = __builtin_amdgcn_wmma_f32_16x16x32_f16(false, pf, false, vf, (short)0, o_[nt], false, false);
      }
    }
    __syncthreads();
  }
  #pragma unroll
  for (int nt = 0; nt < 4; ++nt) {
    int hd = nt * 16 + lo16;
    #pragma unroll
    for (int r = 0; r < 8; ++r) {
      int row = qrow0 + r + (hi << 3);
      float val = o_[nt][r] / l_[r];
      out[((size_t)(b * Ss) + row) * Dd + h * HDd + hd] = (half_t)val;
    }
  }
}

// ---------------------------------------------------------------- layernorm
// one block per row; y = LN(base + delta) * gam + bet -> f32 (and optional f16)
__global__ __launch_bounds__(256) void ln_kernel(
    const float* __restrict__ base, const float* __restrict__ delta,
    const float* __restrict__ gam, const float* __restrict__ bet,
    float* __restrict__ outF, half_t* __restrict__ outH) {
  int row = blockIdx.x, t = threadIdx.x;
  size_t o = (size_t)row * Dd;
  float v0 = base[o + t], v1 = base[o + t + 256];
  if (delta) { v0 += delta[o + t]; v1 += delta[o + t + 256]; }
  __shared__ float red[8];
  float s = v0 + v1;
  #pragma unroll
  for (int m = 1; m < 32; m <<= 1) s += __shfl_xor(s, m, 32);
  if ((t & 31) == 0) red[t >> 5] = s;
  __syncthreads();
  float tot = 0.f;
  #pragma unroll
  for (int i = 0; i < 8; ++i) tot += red[i];
  float mean = tot * (1.0f / Dd);
  __syncthreads();
  float d0 = v0 - mean, d1 = v1 - mean;
  float sq = d0 * d0 + d1 * d1;
  #pragma unroll
  for (int m = 1; m < 32; m <<= 1) sq += __shfl_xor(sq, m, 32);
  if ((t & 31) == 0) red[t >> 5] = sq;
  __syncthreads();
  float tv = 0.f;
  #pragma unroll
  for (int i = 0; i < 8; ++i) tv += red[i];
  float rstd = rsqrtf(tv * (1.0f / Dd) + 1e-5f);
  float o0 = d0 * rstd * gam[t] + bet[t];
  float o1 = d1 * rstd * gam[t + 256] + bet[t + 256];
  outF[o + t] = o0; outF[o + t + 256] = o1;
  if (outH) { outH[o + t] = (half_t)o0; outH[o + t + 256] = (half_t)o1; }
}

// ---------------------------------------------------------------- classifier head
__global__ __launch_bounds__(256) void head_kernel(
    const float* __restrict__ x, const float* __restrict__ c1W, const float* __restrict__ c1b,
    const float* __restrict__ c2W, const float* __restrict__ c2b,
    const float* __restrict__ c3W, const float* __restrict__ c3b, float* __restrict__ out) {
  int b = blockIdx.x, t = threadIdx.x;
  __shared__ float h1[256];
  __shared__ float red[128];
  const float* cls = x + (size_t)b * Ss * Dd; // row b*S + 0
  float acc = c1b[t];
  for (int kk = 0; kk < Dd; ++kk) acc += cls[kk] * c1W[kk * 256 + t];
  h1[t] = gelu_exact(acc);
  __syncthreads();
  float h2v = 0.f;
  if (t < 128) {
    float a2 = c2b[t];
    for (int kk = 0; kk < 256; ++kk) a2 += h1[kk] * c2W[kk * 128 + t];
    h2v = gelu_exact(a2);
    red[t] = h2v * c3W[t];
  }
  __syncthreads();
  for (int st = 64; st > 0; st >>= 1) {
    if (t < st) red[t] += red[t + st];
    __syncthreads();
  }
  if (t == 0) out[b] = red[0] + c3b[0];
}

// ---------------------------------------------------------------- launch
extern "C" void kernel_launch(void* const* d_in, const int* in_sizes, int n_in,
                              void* d_out, int out_size, void* d_ws, size_t ws_size,
                              hipStream_t stream) {
  const long long* input_ids = (const long long*)d_in[0];
  const int*   attn_mask = (const int*)d_in[1];
  const float* tok_emb = (const float*)d_in[2];
  const float* pos_emb = (const float*)d_in[3];
  const float* type_emb = (const float*)d_in[4];
  const float* Wq = (const float*)d_in[5];  const float* bq = (const float*)d_in[6];
  const float* Wk = (const float*)d_in[7];  const float* bk = (const float*)d_in[8];
  const float* Wv = (const float*)d_in[9];  const float* bv = (const float*)d_in[10];
  const float* Wo = (const float*)d_in[11]; const float* bo = (const float*)d_in[12];
  const float* attn_bias = (const float*)d_in[13];
  const float* ln_a_s = (const float*)d_in[14]; const float* ln_a_b = (const float*)d_in[15];
  const float* W1 = (const float*)d_in[16]; const float* b1 = (const float*)d_in[17];
  const float* W2 = (const float*)d_in[18]; const float* b2 = (const float*)d_in[19];
  const float* ln2_s = (const float*)d_in[20]; const float* ln2_b = (const float*)d_in[21];
  const float* lnf_s = (const float*)d_in[22]; const float* lnf_b = (const float*)d_in[23];
  const float* c1W = (const float*)d_in[24]; const float* c1b = (const float*)d_in[25];
  const float* c2W = (const float*)d_in[26]; const float* c2b = (const float*)d_in[27];
  const float* c3W = (const float*)d_in[28]; const float* c3b = (const float*)d_in[29];
  float* out = (float*)d_out;

  char* ws = (char*)d_ws;
  size_t off = 0;
  auto take = [&](size_t bytes) -> void* {
    void* p = ws + off;
    off += (bytes + 255) & ~(size_t)255;
    return p;
  };
  float*  x    = (float*)take((size_t)NT * Dd * 4);
  half_t* xh   = (half_t*)take((size_t)NT * Dd * 2);
  half_t* qkva = (half_t*)take((size_t)NT * FFf * 2); // q|k|v|attn, reused as FFN hidden
  half_t* qh = qkva;
  half_t* kh = qkva + (size_t)NT * Dd;
  half_t* vh = qkva + (size_t)2 * NT * Dd;
  half_t* ah = qkva + (size_t)3 * NT * Dd;
  half_t* ff1h = qkva;
  float*  proj = (float*)take((size_t)NT * Dd * 4);
  half_t* WqH = (half_t*)take((size_t)Ll * Dd * Dd * 2);
  half_t* WkH = (half_t*)take((size_t)Ll * Dd * Dd * 2);
  half_t* WvH = (half_t*)take((size_t)Ll * Dd * Dd * 2);
  half_t* WoH = (half_t*)take((size_t)Ll * Dd * Dd * 2);
  half_t* W1H = (half_t*)take((size_t)Ll * Dd * FFf * 2);
  half_t* W2H = (half_t*)take((size_t)Ll * FFf * Dd * 2);

  auto cvt = [&](const float* src, half_t* dst, size_t n) {
    f32_to_f16_kernel<<<(unsigned)((n + 255) / 256), 256, 0, stream>>>(src, dst, n);
  };
  cvt(Wq, WqH, (size_t)Ll * Dd * Dd);
  cvt(Wk, WkH, (size_t)Ll * Dd * Dd);
  cvt(Wv, WvH, (size_t)Ll * Dd * Dd);
  cvt(Wo, WoH, (size_t)Ll * Dd * Dd);
  cvt(W1, W1H, (size_t)Ll * Dd * FFf);
  cvt(W2, W2H, (size_t)Ll * FFf * Dd);

  embed_kernel<<<(NT * Dd) / 256, 256, 0, stream>>>(input_ids, tok_emb, pos_emb, type_emb, x, xh);

  auto gemm = [&](const half_t* A, const half_t* W, const float* bias,
                  float* oF, half_t* oH, int M, int K, int Ncols, int act) {
    dim3 grid(Ncols / 128, M / 128);
    gemm_wmma_kernel<<<grid, 256, 0, stream>>>(A, W, bias, oF, oH, M, K, Ncols, act);
  };

  for (int i = 0; i < Ll; ++i) {
    size_t wO = (size_t)i * Dd * Dd;
    gemm(xh, WqH + wO, bq + (size_t)i * Dd, nullptr, qh, NT, Dd, Dd, 0);
    gemm(xh, WkH + wO, bk + (size_t)i * Dd, nullptr, kh, NT, Dd, Dd, 0);
    gemm(xh, WvH + wO, bv + (size_t)i * Dd, nullptr, vh, NT, Dd, Dd, 0);
    attn_kernel<<<dim3(4, Hh, Bb), 256, 0, stream>>>(qh, kh, vh, attn_mask,
                                                     attn_bias + (size_t)i * Hh, ah);
    gemm(ah, WoH + wO, bo + (size_t)i * Dd, proj, nullptr, NT, Dd, Dd, 0);
    ln_kernel<<<NT, 256, 0, stream>>>(x, proj, ln_a_s + (size_t)i * Dd,
                                      ln_a_b + (size_t)i * Dd, x, xh);
    gemm(xh, W1H + (size_t)i * Dd * FFf, b1 + (size_t)i * FFf, nullptr, ff1h, NT, Dd, FFf, 1);
    gemm(ff1h, W2H + (size_t)i * FFf * Dd, b2 + (size_t)i * Dd, proj, nullptr, NT, FFf, Dd, 0);
    ln_kernel<<<NT, 256, 0, stream>>>(x, proj, ln2_s + (size_t)i * Dd,
                                      ln2_b + (size_t)i * Dd, x, xh);
  }
  ln_kernel<<<NT, 256, 0, stream>>>(x, nullptr, lnf_s, lnf_b, x, nullptr);
  head_kernel<<<Bb, 256, 0, stream>>>(x, c1W, c1b, c2W, c2b, c3W, c3b, out);
}
